// YuanMengAttention_22754736734816
// MI455X (gfx1250) — matrix operand order repaired and checked by reading, not compiled, not run on hardware
//
#include <hip/hip_runtime.h>
#include <hip/hip_bf16.h>

// ---------- sizes (B=1) ----------
#define S 4096
#define D 1024
#define NH 16
#define NKV 4
#define HD 64
#define EQKV 1536   // (NH + 2*NKV) * HD

typedef __attribute__((ext_vector_type(16))) __bf16 v16bf;
typedef __attribute__((ext_vector_type(8)))  __bf16 v8bf;
typedef __attribute__((ext_vector_type(8)))  float  v8f;

__device__ __forceinline__ __bf16 f2bf(float f) { return (__bf16)f; }

// A-fragment (16x32 bf16, row m = lane&15): e<8 -> k=hi*8+e ; e>=8 -> k=16+hi*8+(e-8)
__device__ __forceinline__ v16bf load_a_bf16(const __bf16* __restrict__ row, int hi) {
  const v8bf* p = (const v8bf*)(row + hi * 8);
  v8bf a0 = p[0], a1 = p[2];            // +16 elements
  v16bf a;
#pragma unroll
  for (int e = 0; e < 8; ++e) { a[e] = a0[e]; a[e + 8] = a1[e]; }
  return a;
}

// B-fragment (32x16 bf16, col n = lane&15): e<8 -> k=hi*8+e ; e>=8 -> k=16+hi*8+(e-8) (K-contiguous source)
__device__ __forceinline__ v16bf load_b_bf16(const __bf16* __restrict__ col, int hi) {
  const v8bf* p = (const v8bf*)(col + hi * 8);
  v8bf b0 = p[0], b1 = p[2];
  v16bf b;
#pragma unroll
  for (int e = 0; e < 8; ++e) { b[e] = b0[e]; b[e + 8] = b1[e]; }
  return b;
}

__device__ __forceinline__ v8f wmma_bf16(v16bf a, v16bf b, v8f c) {
  return __builtin_amdgcn_wmma_f32_16x16x32_bf16(false, a, false, b, (short)0, c,
                                                 false, false);
}

// ---------- kernel: f32 -> bf16 elementwise ----------
__global__ void k_cvt_bf16(const float* __restrict__ in, __bf16* __restrict__ out, int n) {
  int i = blockIdx.x * blockDim.x + threadIdx.x;
  if (i < n) out[i] = f2bf(in[i]);
}

// ---------- kernel: W[K][N] f32 -> Wt[N][K] bf16 ----------
__global__ void k_transpose_bf16(const float* __restrict__ W, __bf16* __restrict__ Wt,
                                 int K, int N) {
  int idx = blockIdx.x * blockDim.x + threadIdx.x;
  if (idx >= K * N) return;
  int k = idx / N, n = idx - k * N;
  Wt[(size_t)n * K + k] = f2bf(W[idx]);
}

// ---------- kernel: C[M][N] f32 = A[M][K] bf16  x  Bt[N][K] bf16 ----------
// one wave -> 32x32 output tile (2x2 WMMA fragments), K step 32;
// inner loop is pure b128 loads + v_wmma (matrix-pipe bound)
__global__ void k_gemm_bbt(const __bf16* __restrict__ A, const __bf16* __restrict__ Bt,
                           float* __restrict__ C, int M, int N, int K) {
  int wave = (int)((blockIdx.x * blockDim.x + threadIdx.x) >> 5);
  int lane = threadIdx.x & 31;
  int nl = lane & 15, hi = lane >> 4;
  int tilesN = N >> 5;
  int tm = (wave / tilesN) << 5;
  int tn = (wave % tilesN) << 5;
  if (tm >= M) return;
  v8f acc[2][2] = {};
  for (int k0 = 0; k0 < K; k0 += 32) {
    v16bf a[2], b[2];
#pragma unroll
    for (int i = 0; i < 2; ++i)
      a[i] = load_a_bf16(A + (size_t)(tm + 16 * i + nl) * K + k0, hi);
#pragma unroll
    for (int j = 0; j < 2; ++j)
      b[j] = load_b_bf16(Bt + (size_t)(tn + 16 * j + nl) * K + k0, hi);
#pragma unroll
    for (int i = 0; i < 2; ++i)
#pragma unroll
      for (int j = 0; j < 2; ++j)
        acc[i][j] = wmma_bf16(a[i], b[j], acc[i][j]);
  }
#pragma unroll
  for (int i = 0; i < 2; ++i)
#pragma unroll
    for (int j = 0; j < 2; ++j)
#pragma unroll
      for (int r = 0; r < 8; ++r)
        C[(size_t)(tm + 16 * i + r + (hi << 3)) * N + tn + 16 * j + nl] = acc[i][j][r];
}

// ---------- kernel: RoPE + split + layout ----------
// QKV[s][1536] f32 -> Qb[NH][S][HD] bf16 (roped, PRE-SCALED by log2(e)/sqrt(HD)
//                     so attention runs in the base-2 domain -> raw v_exp_f32),
//                     Kb[NKV][S][HD] bf16 (roped),
//                     Vt[NKV][HD][S] bf16 (transposed for P.V B-fragments)
__global__ void k_rope_split(const float* __restrict__ QKV,
                             const float* __restrict__ pcos, const float* __restrict__ psin,
                             __bf16* __restrict__ Qb, __bf16* __restrict__ Kb,
                             __bf16* __restrict__ Vt) {
  int tid = blockIdx.x * blockDim.x + threadIdx.x;
  if (tid >= S * (EQKV / 2)) return;
  int s = tid / (EQKV / 2);
  int p = tid - s * (EQKV / 2);
  int e = p * 2;
  float te = QKV[(size_t)s * EQKV + e];
  float to = QKV[(size_t)s * EQKV + e + 1];
  if (e < (NH + NKV) * HD) {                         // q or k: apply RoPE
    int c = (e < NH * HD) ? e : (e - NH * HD);
    int hh = c >> 6;
    int d  = c & 63;
    int i  = d >> 1;
    float cs = pcos[s * (HD / 2) + i];
    float sn = psin[s * (HD / 2) + i];
    float oe = te * cs - to * sn;
    float oo = te * sn + to * cs;
    if (e < NH * HD) {                               // q: fold softmax scale * log2(e)
      const float qs = 0.125f * 1.44269504088896f;
      __bf16* dst = Qb + ((size_t)hh * S + s) * HD + d;
      dst[0] = f2bf(oe * qs);
      dst[1] = f2bf(oo * qs);
    } else {
      __bf16* dst = Kb + ((size_t)hh * S + s) * HD + d;
      dst[0] = f2bf(oe);
      dst[1] = f2bf(oo);
    }
  } else {                                           // v: transpose into [h][d][s]
    int c = e - (NH + NKV) * HD;
    int hh = c >> 6;
    int d  = c & 63;
    Vt[((size_t)hh * HD + d) * S + s]       = f2bf(te);
    Vt[((size_t)hh * HD + d + 1) * S + s]   = f2bf(to);
  }
}

// ---------- kernel: causal flash attention ----------
// one wave per (head, 16-query tile); key blocks of 32; online softmax in base-2.
// Row sums come FREE from an extra WMMA against an all-ones B fragment
// (every output column equals the row sum -> every lane holds it, no shuffles).
// Key loop split into unmasked-main + masked-diagonal to keep the hot loop lean.
__global__ void k_flash_attn(const __bf16* __restrict__ Qb, const __bf16* __restrict__ Kb,
                             const __bf16* __restrict__ Vt, __bf16* __restrict__ O) {
  __shared__ __attribute__((aligned(16))) __bf16 plds[4][16][32];
  int lane = threadIdx.x & 31;
  int wv = threadIdx.x >> 5;
  int wave = blockIdx.x * 4 + wv;
  int h  = wave >> 8;            // 16 heads
  int qt = wave & 255;           // 256 query tiles of 16
  int q0 = qt << 4;
  int hk = h >> 2;               // GQA: 4 q-heads per kv-head
  int nl = lane & 15, hi = lane >> 4;

  const __bf16* Kbase = Kb + (size_t)hk * S * HD;
  const __bf16* Vbase = Vt + (size_t)hk * HD * S;

  v16bf qf[2];
#pragma unroll
  for (int t = 0; t < 2; ++t)
    qf[t] = load_a_bf16(Qb + ((size_t)h * S + q0 + nl) * HD + t * 32, hi);

  v16bf ones;
#pragma unroll
  for (int e = 0; e < 16; ++e) ones[e] = (__bf16)1.0f;

  float mrun[8];
#pragma unroll
  for (int r = 0; r < 8; ++r) mrun[r] = -3.0e38f;
  v8f o[4] = {};
  v8f l4 = {};                   // running row sums (ones-column accumulator)

  auto process = [&](int j0, bool domask) {
    v8f s0 = {}, s1 = {};
#pragma unroll
    for (int t = 0; t < 2; ++t) {             // S = Q(16x64) . K^T(64x32)
      v16bf k0f = load_b_bf16(Kbase + (size_t)(j0 + nl) * HD + t * 32, hi);
      v16bf k1f = load_b_bf16(Kbase + (size_t)(j0 + 16 + nl) * HD + t * 32, hi);
      s0 = wmma_bf16(qf[t], k0f, s0);
      s1 = wmma_bf16(qf[t], k1f, s1);
    }
    float bm[8];
    if (domask) {
#pragma unroll
      for (int r = 0; r < 8; ++r) {
        int mg = q0 + r + (hi << 3);
        float v0 = (j0 + nl      <= mg) ? s0[r] : -1.0e9f;
        float v1 = (j0 + 16 + nl <= mg) ? s1[r] : -1.0e9f;
        s0[r] = v0; s1[r] = v1;
        bm[r] = fmaxf(v0, v1);
      }
    } else {
#pragma unroll
      for (int r = 0; r < 8; ++r) bm[r] = fmaxf(s0[r], s1[r]);
    }
#pragma unroll
    for (int off = 1; off < 16; off <<= 1)    // row max within 16-lane half
#pragma unroll
      for (int r = 0; r < 8; ++r)
        bm[r] = fmaxf(bm[r], __shfl_xor(bm[r], off, 32));
#pragma unroll
    for (int r = 0; r < 8; ++r) {             // base-2 online rescale + exponentials
      float mn = fmaxf(mrun[r], bm[r]);
      float alpha = __builtin_amdgcn_exp2f(mrun[r] - mn);   // raw v_exp_f32
      mrun[r] = mn;
      s0[r] = __builtin_amdgcn_exp2f(s0[r] - mn);
      s1[r] = __builtin_amdgcn_exp2f(s1[r] - mn);
      l4[r] *= alpha;
#pragma unroll
      for (int f = 0; f < 4; ++f) o[f][r] *= alpha;
    }
    // P: C-layout -> LDS -> A-layout (same-wave LDS ops are in-order)
#pragma unroll
    for (int r = 0; r < 8; ++r) {
      plds[wv][r + (hi << 3)][nl]      = f2bf(s0[r]);
      plds[wv][r + (hi << 3)][nl + 16] = f2bf(s1[r]);
    }
    asm volatile("" ::: "memory");
    v16bf pf = load_a_bf16(&plds[wv][nl][0], hi);
    asm volatile("" ::: "memory");
#pragma unroll
    for (int f = 0; f < 4; ++f) {             // O += P(16x32) . V(32x64)
      v16bf vf = load_b_bf16(Vbase + (size_t)(f * 16 + nl) * S + j0, hi);
      o[f] = wmma_bf16(pf, vf, o[f]);
    }
    l4 = wmma_bf16(pf, ones, l4);             // row sums via matrix pipe
  };

  int jdiag = q0 >> 5;                        // blocks [0,jdiag) need no mask
  int jend  = (q0 + 15) >> 5;
  for (int jb = 0; jb < jdiag; ++jb) process(jb << 5, false);
  for (int jb = jdiag; jb <= jend; ++jb) process(jb << 5, true);

#pragma unroll
  for (int f = 0; f < 4; ++f)
#pragma unroll
    for (int r = 0; r < 8; ++r) {
      float inv = __builtin_amdgcn_rcpf(l4[r]);
      O[(size_t)(q0 + r + (hi << 3)) * (NH * HD) + h * HD + f * 16 + nl] =
          f2bf(o[f][r] * inv);
    }
}

// ---------- host ----------
extern "C" void kernel_launch(void* const* d_in, const int* in_sizes, int n_in,
                              void* d_out, int out_size, void* d_ws, size_t ws_size,
                              hipStream_t stream) {
  const float* x     = (const float*)d_in[0];
  const float* pcos  = (const float*)d_in[1];
  const float* psin  = (const float*)d_in[2];
  const float* w_qkv = (const float*)d_in[3];
  const float* w_o   = (const float*)d_in[4];
  float* out = (float*)d_out;

  char* ws = (char*)d_ws;
  size_t off = 0;
  __bf16* WqkvT = (__bf16*)(ws + off); off += (size_t)EQKV * D * 2;        // 3.0 MB
  __bf16* WoT   = (__bf16*)(ws + off); off += (size_t)D * D * 2;           // 2.0 MB
  __bf16* Xb    = (__bf16*)(ws + off); off += (size_t)S * D * 2;           // 8.0 MB
  float*  QKV   = (float*)(ws + off);  off += (size_t)S * EQKV * 4;        // 24 MB
  __bf16* Qb    = (__bf16*)(ws + off); off += (size_t)NH * S * HD * 2;     // 8 MB
  __bf16* Kb    = (__bf16*)(ws + off); off += (size_t)NKV * S * HD * 2;    // 2 MB
  __bf16* Vt    = (__bf16*)(ws + off); off += (size_t)NKV * HD * S * 2;    // 2 MB
  __bf16* Obf   = (__bf16*)(ws + off); off += (size_t)S * NH * HD * 2;     // 8 MB

  // 1) precision prep (bandwidth-trivial)
  k_cvt_bf16<<<(S * D + 255) / 256, 256, 0, stream>>>(x, Xb, S * D);
  k_transpose_bf16<<<(D * EQKV + 255) / 256, 256, 0, stream>>>(w_qkv, WqkvT, D, EQKV);
  k_transpose_bf16<<<(D * D + 255) / 256, 256, 0, stream>>>(w_o, WoT, D, D);

  // 2) QKV projection: (S/32)*(EQKV/32) = 6144 waves -> 768 blocks of 256
  k_gemm_bbt<<<768, 256, 0, stream>>>(Xb, WqkvT, QKV, S, EQKV, D);

  // 3) RoPE + split
  k_rope_split<<<(S * (EQKV / 2) + 255) / 256, 256, 0, stream>>>(QKV, pcos, psin,
                                                                 Qb, Kb, Vt);

  // 4) flash attention: 16 heads * 256 q-tiles = 4096 waves -> 1024 blocks of 128
  k_flash_attn<<<1024, 128, 0, stream>>>(Qb, Kb, Vt, Obf);

  // 5) output projection: (S/32)*(D/32) = 4096 waves -> 512 blocks of 256
  k_gemm_bbt<<<512, 256, 0, stream>>>(Obf, WoT, out, S, D, D);
}
